// ValueHead_27625229648505
// MI455X (gfx1250) — matrix operand order repaired and checked by reading, not compiled, not run on hardware
//
#include <hip/hip_runtime.h>

// ============================================================================
// ValueHead fused pipeline for MI455X (gfx1250, wave32, WMMA bf16)
//
// Math collapse:
//   scores = x @ S + c          (S = Wk^T q precomputed, 256x8)
//   [v|scores] = hcat @ Bcomb   (Bcomb = merge^T @ [Wv^T | S], 512x264)
//   out[g] = tanh(pooled . wfin + bfin)
//   BN: linear bias cancels -> h = relu(y_nobias*alpha + beta2)
//
// Pass structure (all on `stream`, deterministic, no atomics):
//   k_prep1 / k_conv / k_bcomb / k_bb : tiny device-side weight algebra
//   k_stats  : GEMM1 (WMMA) -> per-WG column sum/sumsq partials (400 WGs)
//   k_red    : deterministic reduction of partials -> stats
//   k_norm   : stats -> alpha/beta2
//   k_mega   : per group (100 rows): GEMM1 + BN/ReLU -> h (LDS bf16)
//              -> GEMM2 scores -> softmax -> GEMM2 v + pooling -> out[g]
// Workspace use: ~2.2 MB.
// ============================================================================

typedef unsigned short u16;
typedef __attribute__((ext_vector_type(16))) __bf16 v16bf;
typedef __attribute__((ext_vector_type(8)))  float  v8f;

__device__ inline u16 f2bf(float f) {
  unsigned u = __float_as_uint(f);
  u = (u + 0x7FFFu + ((u >> 16) & 1u)) >> 16;   // round-to-nearest-even
  return (u16)u;
}

struct B32x { uint4 a, b; };
__device__ inline v16bf mkv(uint4 lo, uint4 hi) {
  union { B32x s; v16bf v; } u;
  u.s.a = lo; u.s.b = hi;
  return u.v;
}

// A-tile (16x32 bf16) from row-major [m][k] storage, stride `strd` elems.
// ISA layout: lanes 0-15: M=lane, K={0..7,16..23}; lanes 16-31: M=lane-16,
// K={8..15,24..31}. Two 16B chunks per lane.
__device__ inline v16bf ldA(const u16* base, int strd, int lane) {
  int m  = lane & 15;
  int k0 = (lane < 16) ? 0 : 8;
  const u16* p = base + m * strd + k0;
  uint4 lo = *(const uint4*)(p);
  uint4 hi = *(const uint4*)(p + 16);
  return mkv(lo, hi);
}

// B-tile (32x16 bf16) from N-major [n][k] storage, stride `strd` elems.
// ISA layout: lanes 0-15: N=lane, K={0..7,16..23}; lanes 16-31: N=lane-16,
// K={8..15,24..31}. Two 16B chunks per lane.
__device__ inline v16bf ldB(const u16* base, int strd, int lane) {
  int n  = lane & 15;
  int k0 = (lane < 16) ? 0 : 8;
  const u16* p = base + n * strd + k0;
  uint4 lo = *(const uint4*)(p);
  uint4 hi = *(const uint4*)(p + 16);
  return mkv(lo, hi);
}

__device__ inline v8f wmma_bf16(v16bf a, v16bf b, v8f c) {
  return __builtin_amdgcn_wmma_f32_16x16x32_bf16(false, a, false, b,
                                                 (short)0, c, false, false);
}

// ---------------------------------------------------------------------------
// Prologue: q, S, c, wfin, bfin (single block)
// ---------------------------------------------------------------------------
__global__ void k_prep1(const float* inW, const float* inb,
                        const float* outW, const float* outb,
                        const float* rW, const float* rb,
                        float* qv, float* Sarr, float* cvec,
                        float* wfin, float* bfin) {
  int tid = threadIdx.x; // 256 threads
  { // q[hd] = (rowsum(Wq[hd]) + bq[hd]) / sqrt(32)
    float acc = 0.f;
    const float* row = inW + tid * 256;
    for (int i = 0; i < 256; ++i) acc += row[i];
    qv[tid] = (acc + inb[tid]) * 0.17677669529663687f;
  }
  __syncthreads();
  for (int idx = tid; idx < 2048; idx += 256) { // S[j][h]
    int j = idx >> 3, h = idx & 7;
    float acc = 0.f;
    for (int d = 0; d < 32; ++d)
      acc += qv[h * 32 + d] * inW[(256 + h * 32 + d) * 256 + j];
    Sarr[j * 8 + h] = acc;
  }
  if (tid < 8) { // c[h]
    float acc = 0.f;
    for (int d = 0; d < 32; ++d) acc += qv[tid * 32 + d] * inb[256 + tid * 32 + d];
    cvec[tid] = acc;
  }
  { // wfin[o] = sum_j readout[j] * outW[j][o]
    float acc = 0.f;
    for (int j = 0; j < 256; ++j) acc += rW[j] * outW[j * 256 + tid];
    wfin[tid] = acc;
  }
  if (tid == 0) {
    float acc = 0.f;
    for (int j = 0; j < 256; ++j) acc += outb[j] * rW[j];
    bfin[0] = acc + rb[0];
  }
}

// Convert p0_W / p1_W to bf16 (row-major = N-major for GEMM1 B-tiles).
__global__ void k_conv(const float* W0, const float* W1, u16* W0b, u16* W1b) {
  int idx = blockIdx.x * blockDim.x + threadIdx.x; // 131072
  if (idx < 65536) W0b[idx] = f2bf(W0[idx]);
  else             W1b[idx - 65536] = f2bf(W1[idx - 65536]);
}

// Bcomb[o][i] (o-major, 272x512, rows 264..271 zero-padded), bf16.
__global__ void k_bcomb(const float* mW, const float* inW, const float* Sarr,
                        u16* Bc) {
  int idx = blockIdx.x * blockDim.x + threadIdx.x; // 272*512 = 139264
  if (idx >= 272 * 512) return;
  int o = idx >> 9, i = idx & 511;
  float acc = 0.f;
  if (o < 256) {
    const float* R = inW + (512 + o) * 256;               // Wv row o
    for (int j = 0; j < 256; ++j) acc += mW[j * 512 + i] * R[j];
  } else if (o < 264) {
    int h = o - 256;
    for (int j = 0; j < 256; ++j) acc += mW[j * 512 + i] * Sarr[j * 8 + h];
  }
  Bc[o * 512 + i] = f2bf(acc);
}

// Combined bias bb[272].
__global__ void k_bb(const float* mb, const float* inW, const float* inb,
                     const float* Sarr, const float* cvec, float* bb) {
  int o = blockIdx.x * blockDim.x + threadIdx.x;
  if (o >= 272) return;
  float acc = 0.f;
  if (o < 256) {
    for (int j = 0; j < 256; ++j) acc += mb[j] * inW[(512 + o) * 256 + j];
    acc += inb[512 + o];
  } else if (o < 264) {
    int h = o - 256;
    for (int j = 0; j < 256; ++j) acc += mb[j] * Sarr[j * 8 + h];
    acc += cvec[h];
  }
  bb[o] = acc;
}

// ---------------------------------------------------------------------------
// Stats pass: y = x @ W^T (no bias) via WMMA; per-WG column sum/sumsq.
// 400 WGs x 512 rows (4 chunks of 128). Deterministic: partials to ws.
// ---------------------------------------------------------------------------
__global__ void __launch_bounds__(256)
k_stats(const float* x0, const float* x1,
        const u16* W0b, const u16* W1b, float* partial) {
  extern __shared__ char smem[];
  u16* xs = (u16*)smem; // 128 x 256 bf16 = 64KB
  int tid = threadIdx.x, wave = tid >> 5, lane = tid & 31;
  float* pout = partial + (size_t)blockIdx.x * 1024;

  for (int mat = 0; mat < 2; ++mat) {
    const float* x  = mat ? x1 : x0;
    const u16*   Wb = mat ? W1b : W0b;
    float ssum[2] = {0.f, 0.f}, ssq[2] = {0.f, 0.f};
    for (int chunk = 0; chunk < 4; ++chunk) {
      int rowbase = blockIdx.x * 512 + chunk * 128;
      __syncthreads();
      for (int i = tid; i < 128 * 64; i += 256) {
        int rr = i >> 6, c4 = (i & 63) << 2;
        float4 q4 = *(const float4*)(x + (size_t)(rowbase + rr) * 256 + c4);
        u16* d = xs + rr * 256 + c4;
        d[0] = f2bf(q4.x); d[1] = f2bf(q4.y); d[2] = f2bf(q4.z); d[3] = f2bf(q4.w);
      }
      __syncthreads();
      for (int sub = 0; sub < 2; ++sub) {
        int nt = wave * 2 + sub;
        v16bf Bk[8];
        for (int kt = 0; kt < 8; ++kt)
          Bk[kt] = ldB(Wb + nt * 16 * 256 + kt * 32, 256, lane);
        for (int mt = 0; mt < 8; ++mt) {
          v8f acc = {};
          for (int kt = 0; kt < 8; ++kt) {
            v16bf A = ldA(xs + mt * 16 * 256 + kt * 32, 256, lane);
            acc = wmma_bf16(A, Bk[kt], acc);
          }
          for (int r = 0; r < 8; ++r) {
            float y = acc[r];
            ssum[sub] += y; ssq[sub] += y * y;
          }
        }
      }
    }
    for (int sub = 0; sub < 2; ++sub) {
      float ts = ssum[sub] + __shfl_xor(ssum[sub], 16, 32);
      float tq = ssq[sub]  + __shfl_xor(ssq[sub],  16, 32);
      if (lane < 16) {
        int col = (wave * 2 + sub) * 16 + lane;
        pout[mat * 512 + col]       = ts;
        pout[mat * 512 + 256 + col] = tq;
      }
    }
  }
}

// Deterministic reduction of 400 partials.
__global__ void k_red(const float* partial, float* stats) {
  int idx = blockIdx.x * blockDim.x + threadIdx.x; // 1024
  float acc = 0.f;
  for (int w = 0; w < 400; ++w) acc += partial[(size_t)w * 1024 + idx];
  stats[idx] = acc;
}

// alpha/beta2 per column. Linear bias cancels: beta2 = beta - alpha*mean(xW).
__global__ void k_norm(const float* stats, const float* g0, const float* be0,
                       const float* g1, const float* be1,
                       float* alpha, float* beta2) {
  int j = blockIdx.x * blockDim.x + threadIdx.x; // 512
  if (j >= 512) return;
  int mat = j >> 8, col = j & 255;
  float s  = stats[mat * 512 + col];
  float sq = stats[mat * 512 + 256 + col];
  float inv = 1.f / 204800.f;
  float mc  = s * inv;
  float var = sq * inv - mc * mc;
  float a = (mat ? g1[col] : g0[col]) * rsqrtf(var + 1e-5f);
  alpha[j] = a;
  beta2[j] = (mat ? be1[col] : be0[col]) - a * mc;
}

// ---------------------------------------------------------------------------
// Mega fused pass: one WG per group of 100 rows (padded to 112 = 7 M-tiles).
// 16 waves. LDS: xs(112x256 bf16) + hs(112x512 bf16) + softmax/pool buffers.
// ---------------------------------------------------------------------------
__global__ void __launch_bounds__(512)
k_mega(const float* x0, const float* x1,
       const u16* W0b, const u16* W1b, const u16* Bc, const float* bb,
       const float* alpha, const float* beta2,
       const float* wfin, const float* bfin, float* out) {
  extern __shared__ char smem[];
  u16*   xs     = (u16*)smem;                          // 57344 B
  u16*   hs     = (u16*)(smem + 57344);                // 114688 B
  float* e_sc   = (float*)(smem + 57344 + 114688);     // 112*8 f32
  float* den    = e_sc + 112 * 8;                      // 8
  float* pooled = den + 8;                             // 256

  int tid = threadIdx.x, wave = tid >> 5, lane = tid & 31;
  int g = blockIdx.x;
  size_t row0 = (size_t)g * 100;

  // ---- GEMM1 per input branch: y = x@W^T, BN+ReLU -> hs bf16 ----
  for (int mat = 0; mat < 2; ++mat) {
    const float* x  = mat ? x1 : x0;
    const u16*   Wb = mat ? W1b : W0b;
    __syncthreads(); // xs reuse safe
    for (int i = tid; i < 112 * 64; i += 512) {
      int rr = i >> 6, c4 = (i & 63) << 2;
      u16* d = xs + rr * 256 + c4;
      if (rr < 100) {
        float4 q4 = *(const float4*)(x + (row0 + rr) * 256 + c4);
        d[0] = f2bf(q4.x); d[1] = f2bf(q4.y); d[2] = f2bf(q4.z); d[3] = f2bf(q4.w);
      } else { d[0] = 0; d[1] = 0; d[2] = 0; d[3] = 0; }
    }
    __syncthreads();
    {
      int nt = wave; // 16 waves = 16 N-tiles
      v16bf Bk[8];
      for (int kt = 0; kt < 8; ++kt)
        Bk[kt] = ldB(Wb + nt * 16 * 256 + kt * 32, 256, lane);
      int col = mat * 256 + nt * 16 + (lane & 15);
      float al = alpha[col], b2 = beta2[col];
      int rbase = (lane < 16) ? 0 : 8;
      for (int mt = 0; mt < 7; ++mt) {
        v8f acc = {};
        for (int kt = 0; kt < 8; ++kt) {
          v16bf A = ldA(xs + mt * 16 * 256 + kt * 32, 256, lane);
          acc = wmma_bf16(A, Bk[kt], acc);
        }
        for (int r = 0; r < 8; ++r) {
          float h = acc[r] * al + b2;
          h = h > 0.f ? h : 0.f;
          hs[(mt * 16 + rbase + r) * 512 + col] = f2bf(h);
        }
      }
    }
  }
  __syncthreads();

  // ---- GEMM2 score columns (Bcomb rows 256..271, top 8 zero-padded) ----
  if (wave < 7) {
    int mt = wave;
    v8f acc = {};
    for (int kt = 0; kt < 16; ++kt) {
      v16bf A = ldA(hs + mt * 16 * 512 + kt * 32, 512, lane);
      v16bf B = ldB(Bc + 256 * 512 + kt * 32, 512, lane);
      acc = wmma_bf16(A, B, acc);
    }
    int colh = lane & 15;
    if (colh < 8) {
      float bs = bb[256 + colh];
      int rbase = (lane < 16) ? 0 : 8;
      for (int r = 0; r < 8; ++r)
        e_sc[(mt * 16 + rbase + r) * 8 + colh] = acc[r] + bs;
    }
  }
  __syncthreads();

  // ---- segment softmax (per head), wave w = head w ----
  if (wave < 8) {
    int h = wave;
    float mymax = -3.4e38f;
    for (int t = 0; t < 4; ++t) {
      int n = lane + 32 * t;
      if (n < 100) { float v = e_sc[n * 8 + h]; mymax = v > mymax ? v : mymax; }
    }
    for (int m = 16; m >= 1; m >>= 1) {
      float o = __shfl_xor(mymax, m, 32);
      mymax = o > mymax ? o : mymax;
    }
    float mysum = 0.f;
    for (int t = 0; t < 4; ++t) {
      int n = lane + 32 * t;
      if (n < 100) {
        float e = expf(e_sc[n * 8 + h] - mymax);
        e_sc[n * 8 + h] = e; mysum += e;
      } else if (n < 112) {
        e_sc[n * 8 + h] = 0.f; // padded rows contribute nothing
      }
    }
    for (int m = 16; m >= 1; m >>= 1) mysum += __shfl_xor(mysum, m, 32);
    if (lane == 0) den[h] = mysum;
  }
  __syncthreads();

  // ---- GEMM2 v columns fused with pooling: wave = N-tile ----
  {
    int nt = wave;
    int col = nt * 16 + (lane & 15);
    int h = col >> 5;
    float bv = bb[col];
    float pacc = 0.f;
    int rbase = (lane < 16) ? 0 : 8;
    for (int mt = 0; mt < 7; ++mt) {
      v8f acc = {};
      for (int kt = 0; kt < 16; ++kt) {
        v16bf A = ldA(hs + mt * 16 * 512 + kt * 32, 512, lane);
        v16bf B = ldB(Bc + nt * 16 * 512 + kt * 32, 512, lane);
        acc = wmma_bf16(A, B, acc);
      }
      for (int r = 0; r < 8; ++r) {
        int row = mt * 16 + rbase + r;
        pacc += (acc[r] + bv) * e_sc[row * 8 + h];
      }
    }
    pacc += __shfl_xor(pacc, 16, 32);
    if (lane < 16) pooled[col] = pacc / den[h];
  }
  __syncthreads();

  // ---- fused out_proj+readout: out[g] = tanh(pooled . wfin + bfin) ----
  if (wave == 0) {
    float p = 0.f;
    for (int o = lane; o < 256; o += 32) p += pooled[o] * wfin[o];
    for (int m = 16; m >= 1; m >>= 1) p += __shfl_xor(p, m, 32);
    if (lane == 0) out[g] = tanhf(p + bfin[0]);
  }
}

// ---------------------------------------------------------------------------
extern "C" void kernel_launch(void* const* d_in, const int* in_sizes, int n_in,
                              void* d_out, int out_size, void* d_ws, size_t ws_size,
                              hipStream_t stream) {
  const float* x0      = (const float*)d_in[0];
  const float* x1      = (const float*)d_in[1];
  // d_in[2] = batch (int64) unused: segments are contiguous blocks of 100
  const float* p0_W    = (const float*)d_in[3];
  const float* p0_g    = (const float*)d_in[5];
  const float* p0_beta = (const float*)d_in[6];
  const float* p1_W    = (const float*)d_in[7];
  const float* p1_g    = (const float*)d_in[9];
  const float* p1_beta = (const float*)d_in[10];
  const float* mW      = (const float*)d_in[11];
  const float* mb      = (const float*)d_in[12];
  const float* inW     = (const float*)d_in[13];
  const float* inb     = (const float*)d_in[14];
  const float* outW    = (const float*)d_in[15];
  const float* outb    = (const float*)d_in[16];
  const float* rW      = (const float*)d_in[17];
  const float* rb      = (const float*)d_in[18];
  float* out = (float*)d_out;

  // Workspace layout (~2.2 MB total)
  char* w = (char*)d_ws;
  u16*   W0b   = (u16*)(w + 0);        // 131072 B
  u16*   W1b   = (u16*)(w + 131072);   // 131072 B
  u16*   Bc    = (u16*)(w + 262144);   // 272*512*2 = 278528 B
  float* bb    = (float*)(w + 540672); // 272 f32
  float* Sarr  = (float*)(w + 541760); // 2048 f32
  float* cvec  = (float*)(w + 549952); // 8 f32
  float* qv    = (float*)(w + 549984); // 256 f32
  float* wfin  = (float*)(w + 551008); // 256 f32
  float* bfin  = (float*)(w + 552032); // 1 f32
  float* stats = (float*)(w + 552064); // 1024 f32
  float* alpha = (float*)(w + 556160); // 512 f32
  float* beta2 = (float*)(w + 558208); // 512 f32
  float* partial = (float*)(w + 560256); // 400*1024 f32 = 1638400 B

  k_prep1<<<1, 256, 0, stream>>>(inW, inb, outW, outb, rW, rb,
                                 qv, Sarr, cvec, wfin, bfin);
  k_conv <<<512, 256, 0, stream>>>(p0_W, p1_W, W0b, W1b);
  k_bcomb<<<544, 256, 0, stream>>>(mW, inW, Sarr, Bc);
  k_bb   <<<2, 256, 0, stream>>>(mb, inW, inb, Sarr, cvec, bb);
  k_stats<<<400, 256, 65536, stream>>>(x0, x1, W0b, W1b, partial);
  k_red  <<<4, 256, 0, stream>>>(partial, stats);
  k_norm <<<2, 256, 0, stream>>>(stats, p0_g, p0_beta, p1_g, p1_beta, alpha, beta2);

  size_t megaLds = 57344 + 114688 + (size_t)(112 * 8 + 8 + 256) * 4; // 176672
  k_mega <<<2048, 512, megaLds, stream>>>(x0, x1, W0b, W1b, Bc, bb,
                                          alpha, beta2, wfin, bfin, out);
  (void)in_sizes; (void)n_in; (void)out_size; (void)ws_size;
}
